// MathematicalReasoningModel_90202903150964
// MI455X (gfx1250) — matrix operand order repaired and checked by reading, not compiled, not run on hardware
//
#include <hip/hip_runtime.h>

// ---------------------------------------------------------------------------
// MI455X (gfx1250, wave32) fused attention block:
//   out = Attn(RoPE(x Wq^T + bq, x Wk^T + bk), x Wv^T + bv) Wo^T + bo
// All matmuls via v_wmma_f32_16x16x32_bf16 (f32 accumulate).
// Flash-attention K/V tiles staged Global->LDS with async loads (ASYNCcnt).
// Compute-bound problem (~69 GFLOP, ~100MB working set, L2-resident).
// ---------------------------------------------------------------------------

typedef __bf16 bf16_t;
typedef bf16_t v16bf __attribute__((ext_vector_type(16)));
typedef float  v8f   __attribute__((ext_vector_type(8)));

union Frag16 { unsigned int u[8]; v16bf v; };

#define B_DIM  2
#define S_DIM  2048
#define D_DIM  1024
#define H_DIM  16
#define HD_DIM 64

// round-to-nearest-even f32 -> bf16 bits
__device__ __forceinline__ unsigned short f2bf(float f) {
    unsigned int u = __builtin_bit_cast(unsigned int, f);
    u = (u + 0x7FFFu + ((u >> 16) & 1u)) >> 16;
    return (unsigned short)u;
}

// CDNA5 16-bit A-fragment (16x32) per-lane K index for VGPR v, lane-half lh.
__device__ __forceinline__ unsigned afrag_k(unsigned v, unsigned lh) {
    return ((v >> 2) << 4) + (lh << 3) + ((v & 3u) << 1);
}
// CDNA5 16-bit B-fragment (32x16) per-lane K index.
__device__ __forceinline__ unsigned bfrag_k(unsigned v, unsigned lh) {
    return ((v >> 2) << 4) + (lh << 3) + ((v & 3u) << 1);
}

__device__ __forceinline__ v8f wmma_bf16(const Frag16& a, const Frag16& b, v8f c) {
    return __builtin_amdgcn_wmma_f32_16x16x32_bf16(false, a.v, false, b.v,
                                                   (short)0, c, false, false);
}

// Async Global -> LDS copy of 16 bytes per lane (GV addressing, ASYNCcnt).
__device__ __forceinline__ void async_g2l_b128(unsigned lds_addr,
                                               unsigned long long gaddr) {
    asm volatile("global_load_async_to_lds_b128 %0, %1, off"
                 :: "v"(lds_addr), "v"(gaddr) : "memory");
}
__device__ __forceinline__ void wait_asynccnt0() {
    asm volatile("s_wait_asynccnt 0x0" ::: "memory");
}
// low 32 bits of a generic pointer into LDS == LDS byte address (ISA 10.2)
__device__ __forceinline__ unsigned lds_addr_of(const void* p) {
    return (unsigned)(unsigned long long)p;
}

// ---------------------------------------------------------------------------
// elementwise f32 -> bf16
// ---------------------------------------------------------------------------
__global__ void cvt_f32_to_bf16(const float* __restrict__ src,
                                unsigned short* __restrict__ dst, int n) {
    int i = blockIdx.x * blockDim.x + threadIdx.x;
    if (i < n) dst[i] = f2bf(src[i]);
}

// ---------------------------------------------------------------------------
// C[M,N](f32) = A[M,K](bf16) @ W[N,K](bf16)^T + bias[N]
// 256 threads = 8 waves, wave tile 32x64 (2x4 WMMA tiles), WG tile 64x256.
// K stepped by 32. All dims multiples of tile sizes (M=4096,N=1024,K=1024).
// Operands are L2-resident; fragments gathered directly per-lane.
// ---------------------------------------------------------------------------
__global__ __launch_bounds__(256) void gemm_bf16(
        const unsigned short* __restrict__ A,
        const unsigned short* __restrict__ W,
        const float* __restrict__ bias,
        float* __restrict__ C,
        int M, int N, int K) {
    const unsigned lane = threadIdx.x & 31u;
    const unsigned wave = threadIdx.x >> 5;
    const unsigned lm = lane & 15u, lh = lane >> 4;
    const unsigned wm = wave >> 2, wn = wave & 3u;           // 2 x 4 wave grid
    const unsigned mbase = blockIdx.y * 64u + wm * 32u;
    const unsigned nbase = blockIdx.x * 256u + wn * 64u;
    const unsigned int* A32 = (const unsigned int*)A;        // bf16-pair view
    const unsigned int* W32 = (const unsigned int*)W;
    const unsigned uK = (unsigned)K, uN = (unsigned)N;

    v8f acc[2][4];
#pragma unroll
    for (int mt = 0; mt < 2; ++mt)
#pragma unroll
        for (int nt = 0; nt < 4; ++nt) acc[mt][nt] = (v8f){};

    for (unsigned k0 = 0; k0 < uK; k0 += 32u) {
        Frag16 a[2];
#pragma unroll
        for (unsigned mt = 0; mt < 2; ++mt) {
            unsigned row = (mbase + mt * 16u + lm) * uK + k0;
#pragma unroll
            for (unsigned v = 0; v < 8; ++v)
                a[mt].u[v] = A32[(row + afrag_k(v, lh)) >> 1];
        }
        Frag16 bm[4];
#pragma unroll
        for (unsigned nt = 0; nt < 4; ++nt) {
            unsigned row = (nbase + nt * 16u + lm) * uK + k0;
#pragma unroll
            for (unsigned v = 0; v < 8; ++v)
                bm[nt].u[v] = W32[(row + bfrag_k(v, lh)) >> 1];
        }
#pragma unroll
        for (unsigned mt = 0; mt < 2; ++mt)
#pragma unroll
            for (unsigned nt = 0; nt < 4; ++nt)
                acc[mt][nt] = wmma_bf16(a[mt], bm[nt], acc[mt][nt]);
    }

#pragma unroll
    for (unsigned mt = 0; mt < 2; ++mt)
#pragma unroll
        for (unsigned nt = 0; nt < 4; ++nt) {
            unsigned col = nbase + nt * 16u + lm;
            float bv = bias[col];
#pragma unroll
            for (unsigned v = 0; v < 8; ++v) {
                unsigned row = mbase + mt * 16u + lh * 8u + v;  // C layout
                C[row * uN + col] = acc[mt][nt][v] + bv;
            }
        }
}

// ---------------------------------------------------------------------------
// RoPE + reshape [B*S, D](f32) -> [B,H,S,HD](bf16). One thread per rot pair.
// ---------------------------------------------------------------------------
__global__ void rope_reshape(const float* __restrict__ Qf,
                             const float* __restrict__ Kf,
                             unsigned short* __restrict__ Qb,
                             unsigned short* __restrict__ Kb) {
    unsigned tid = blockIdx.x * 256u + threadIdx.x;
    unsigned i = tid & 31u;             // rotation pair index (d=i, d=i+32)
    unsigned h = (tid >> 5) & 15u;
    unsigned s = (tid >> 9) & 2047u;
    unsigned b = tid >> 20;
    // inv_freq = 10000^(-(2i)/64); ln(10000) = 9.210340371976184
    float inv = __expf(-((float)(2u * i) * (1.0f / 64.0f)) * 9.210340371976184f);
    float ang = (float)s * inv;
    float sn, cs;
    __sincosf(ang, &sn, &cs);
    unsigned src = (b * S_DIM + s) * D_DIM + h * HD_DIM;
    unsigned dst = ((b * H_DIM + h) * S_DIM + s) * HD_DIM;
    float q0 = Qf[src + i], q1 = Qf[src + i + 32u];
    float k0 = Kf[src + i], k1 = Kf[src + i + 32u];
    Qb[dst + i]       = f2bf(q0 * cs - q1 * sn);
    Qb[dst + i + 32u] = f2bf(q1 * cs + q0 * sn);
    Kb[dst + i]       = f2bf(k0 * cs - k1 * sn);
    Kb[dst + i + 32u] = f2bf(k1 * cs + k0 * sn);
}

// ---------------------------------------------------------------------------
// V reshape+transpose: [B*S, D](f32) -> Vt[B,H,HD,S](bf16) so that the P@V
// B-fragment's (k,k+1) pair is contiguous in memory (one b32 load per pair).
// ---------------------------------------------------------------------------
__global__ void v_reshape(const float* __restrict__ Vf,
                          unsigned short* __restrict__ Vt) {
    unsigned tid = blockIdx.x * 256u + threadIdx.x;
    unsigned d = tid & 63u;
    unsigned h = (tid >> 6) & 15u;
    unsigned s = (tid >> 10) & 2047u;
    unsigned b = tid >> 21;
    float v = Vf[(b * S_DIM + s) * D_DIM + h * HD_DIM + d];
    Vt[((b * H_DIM + h) * HD_DIM + d) * S_DIM + s] = f2bf(v);
}

// ---------------------------------------------------------------------------
// Flash attention per (b,h): each wave owns 16 query rows, WG owns 128 rows,
// loops over 2048 keys in 64-wide tiles.
//  - K tile (64x64 bf16, 8KB) and V tile (64hd x 64kv bf16, 8KB) are staged
//    Global->LDS once per WG with global_load_async_to_lds_b128 (ASYNCcnt),
//    then all 8 waves read WMMA fragments from LDS (8x fewer global reads).
//  - Scores: 2 chained WMMAs (HD=64); online softmax with __shfl_xor
//    butterflies (masks 1..8 stay inside each 16-lane half = one C-layout row
//    group); P re-laid out via a wave-private LDS tile into A-fragments;
//    P@V accumulated in f32 WMMAs.
// grid = (S/128, B*H), block = 256 (8 waves). LDS: 16KB P + 8KB K + 8KB V.
// ---------------------------------------------------------------------------
__global__ __launch_bounds__(256) void flash_attn(
        const unsigned short* __restrict__ Qb,   // [B,H,S,HD] bf16 (roped)
        const unsigned short* __restrict__ Kb,   // [B,H,S,HD] bf16 (roped)
        const unsigned short* __restrict__ Vt,   // [B,H,HD,S] bf16
        const float* __restrict__ mask,          // [B,S]
        unsigned short* __restrict__ attnb) {    // [B*S, D] bf16
    __shared__ __align__(16) unsigned short plds[8 * 16 * 64];  // 16KB P tiles
    __shared__ __align__(16) unsigned short klds[64 * 64];      // 8KB K tile
    __shared__ __align__(16) unsigned short vlds[64 * 64];      // 8KB V tile

    const unsigned tid = threadIdx.x;
    const unsigned lane = tid & 31u;
    const unsigned wave = tid >> 5;
    const unsigned lm = lane & 15u, lh = lane >> 4;
    const unsigned bh = blockIdx.y;
    const unsigned b = bh >> 4, h = bh & 15u;
    const unsigned qrow0 = blockIdx.x * 128u + wave * 16u;

    const unsigned int* Q32 = (const unsigned int*)Qb;
    const unsigned int* KL32 = (const unsigned int*)klds;
    const unsigned int* VL32 = (const unsigned int*)vlds;
    const float* mrow = mask + b * S_DIM;

    // Staging addresses (per-thread, constant across KV loop except +kv)
    const unsigned klbase = lds_addr_of(klds);
    const unsigned vlbase = lds_addr_of(vlds);
    const unsigned long long kgbase =
        (unsigned long long)Kb + (unsigned long long)(bh * S_DIM) * HD_DIM * 2ull;
    const unsigned long long vgbase = (unsigned long long)Vt;
    const unsigned vcol = bh * 64u;              // Vt HD-row base
    const unsigned vrow0 = tid >> 3;             // 0..31 (V staging row)
    const unsigned vcolb = (tid & 7u) * 16u;     // byte column within V row

    // Q A-fragments for this wave's 16 rows (fixed for whole KV loop)
    Frag16 qa[2];
    {
        unsigned qoff = (bh * S_DIM + qrow0 + lm) * HD_DIM;
#pragma unroll
        for (unsigned ks = 0; ks < 2; ++ks)
#pragma unroll
            for (unsigned v = 0; v < 8; ++v)
                qa[ks].u[v] = Q32[(qoff + ks * 32u + afrag_k(v, lh)) >> 1];
    }

    v8f o[4];
#pragma unroll
    for (int i = 0; i < 4; ++i) o[i] = (v8f){};
    float mrun[8], lrun[8];
#pragma unroll
    for (int v = 0; v < 8; ++v) { mrun[v] = -3.0e38f; lrun[v] = 0.0f; }

    unsigned short* wp = plds + wave * 1024u;      // wave-private 16x64 P tile
    const unsigned int* P32 = (const unsigned int*)wp;

    for (unsigned kv = 0; kv < S_DIM; kv += 64u) {
        // ---- stage K (linear 8KB) and V (64 rows x 128B) tiles into LDS ----
        __syncthreads();   // all waves done reading previous K/V tiles
        {
            unsigned long long kg = kgbase + (unsigned long long)kv * HD_DIM * 2ull;
#pragma unroll
            for (unsigned it = 0; it < 2; ++it) {
                unsigned off = tid * 16u + it * 4096u;
                async_g2l_b128(klbase + off, kg + off);
            }
#pragma unroll
            for (unsigned it = 0; it < 2; ++it) {
                unsigned row = vrow0 + it * 32u;
                unsigned long long vg = vgbase +
                    ((unsigned long long)(vcol + row) * S_DIM + kv) * 2ull + vcolb;
                async_g2l_b128(vlbase + row * 128u + vcolb, vg);
            }
        }
        wait_asynccnt0();
        __syncthreads();   // tiles visible to all waves

        // ---- scores S = (Q K^T) * 1/8 + mask (K fragments from LDS) --------
        v8f sc[4];
#pragma unroll
        for (unsigned nt = 0; nt < 4; ++nt) {
            Frag16 kb0, kb1;
            unsigned krow = (nt * 16u + lm) * 64u;
#pragma unroll
            for (unsigned v = 0; v < 8; ++v) {
                unsigned kk = bfrag_k(v, lh);
                kb0.u[v] = KL32[(krow + kk) >> 1];
                kb1.u[v] = KL32[(krow + 32u + kk) >> 1];
            }
            v8f z = (v8f){};
            z = wmma_bf16(qa[0], kb0, z);
            z = wmma_bf16(qa[1], kb1, z);
            float mval = mrow[kv + nt * 16u + lm];
#pragma unroll
            for (unsigned v = 0; v < 8; ++v)
                sc[nt][v] = z[v] * 0.125f + mval;
        }
        // ---- online softmax: row max over 64 cols (4 tiles x 16 lanes) -----
        float rmax[8];
#pragma unroll
        for (unsigned v = 0; v < 8; ++v) {
            rmax[v] = fmaxf(fmaxf(sc[0][v], sc[1][v]), fmaxf(sc[2][v], sc[3][v]));
#pragma unroll
            for (int off = 1; off <= 8; off <<= 1)
                rmax[v] = fmaxf(rmax[v], __shfl_xor(rmax[v], off, 32));
        }
        float scale[8];
#pragma unroll
        for (unsigned v = 0; v < 8; ++v) {
            float mn = fmaxf(mrun[v], rmax[v]);
            scale[v] = __expf(mrun[v] - mn);
            mrun[v] = mn;
        }
        float rsum[8];
#pragma unroll
        for (unsigned v = 0; v < 8; ++v) rsum[v] = 0.0f;
#pragma unroll
        for (unsigned nt = 0; nt < 4; ++nt)
#pragma unroll
            for (unsigned v = 0; v < 8; ++v) {
                float p = __expf(sc[nt][v] - mrun[v]);
                sc[nt][v] = p;
                rsum[v] += p;
            }
#pragma unroll
        for (unsigned v = 0; v < 8; ++v) {
#pragma unroll
            for (int off = 1; off <= 8; off <<= 1)
                rsum[v] += __shfl_xor(rsum[v], off, 32);
            lrun[v] = lrun[v] * scale[v] + rsum[v];
        }
        // rescale O (lane-local: O C-layout row index == score row index)
#pragma unroll
        for (unsigned nto = 0; nto < 4; ++nto)
#pragma unroll
            for (unsigned v = 0; v < 8; ++v) o[nto][v] *= scale[v];

        // ---- P: C-layout -> LDS (bf16) -> A-fragments ----------------------
#pragma unroll
        for (unsigned nt = 0; nt < 4; ++nt)
#pragma unroll
            for (unsigned v = 0; v < 8; ++v)
                wp[(lh * 8u + v) * 64u + nt * 16u + lm] = f2bf(sc[nt][v]);
        // same-wave DS ops are in-order: safe to read back without barrier
        Frag16 pa[2];
#pragma unroll
        for (unsigned ks = 0; ks < 2; ++ks)
#pragma unroll
            for (unsigned v = 0; v < 8; ++v)
                pa[ks].u[v] = P32[(lm * 64u + ks * 32u + afrag_k(v, lh)) >> 1];

        // ---- O += P @ V (V fragments from LDS) -----------------------------
#pragma unroll
        for (unsigned nto = 0; nto < 4; ++nto) {
#pragma unroll
            for (unsigned ks = 0; ks < 2; ++ks) {
                Frag16 vb;
                unsigned vrow = (nto * 16u + lm) * 64u + ks * 32u;
#pragma unroll
                for (unsigned v = 0; v < 8; ++v)
                    vb.u[v] = VL32[(vrow + bfrag_k(v, lh)) >> 1];
                o[nto] = wmma_bf16(pa[ks], vb, o[nto]);
            }
        }
    }

    // ---- epilogue: O/l -> attn [B*S, H*HD] bf16 ---------------------------
#pragma unroll
    for (unsigned nto = 0; nto < 4; ++nto)
#pragma unroll
        for (unsigned v = 0; v < 8; ++v) {
            unsigned m = lh * 8u + v;
            float val = o[nto][v] / lrun[v];
            unsigned q = qrow0 + m;
            attnb[(b * S_DIM + q) * D_DIM + h * HD_DIM + nto * 16u + lm] =
                f2bf(val);
        }
}

// ---------------------------------------------------------------------------
// Host side: 6-stage pipeline on `stream`. Workspace layout (bytes):
//   xb     [0        , 8388608 )  x bf16
//   wqb    [8388608  , 10485760)  wq bf16
//   wkb    [10485760 , 12582912)
//   wvb    [12582912 , 14680064)
//   wob    [14680064 , 16777216)
//   Qf     [16777216 , 33554432)  f32   (reused as attn bf16 after RoPE)
//   Kf     [33554432 , 50331648)  f32
//   Vf     [50331648 , 67108864)  f32
//   Qb     [67108864 , 75497472)  bf16 roped [B,H,S,HD]
//   Kb     [75497472 , 83886080)  bf16 roped
//   Vt     [83886080 , 92274688)  bf16 [B,H,HD,S]
// Requires ws_size >= 92274688.
// ---------------------------------------------------------------------------
extern "C" void kernel_launch(void* const* d_in, const int* in_sizes, int n_in,
                              void* d_out, int out_size, void* d_ws, size_t ws_size,
                              hipStream_t stream) {
    (void)in_sizes; (void)n_in; (void)out_size; (void)ws_size;
    const float* x    = (const float*)d_in[0];
    const float* wq   = (const float*)d_in[1];
    const float* bq   = (const float*)d_in[2];
    const float* wk   = (const float*)d_in[3];
    const float* bk   = (const float*)d_in[4];
    const float* wv   = (const float*)d_in[5];
    const float* bv   = (const float*)d_in[6];
    const float* wo   = (const float*)d_in[7];
    const float* bo   = (const float*)d_in[8];
    const float* am   = (const float*)d_in[9];

    char* ws = (char*)d_ws;
    unsigned short* xb  = (unsigned short*)(ws);
    unsigned short* wqb = (unsigned short*)(ws + 8388608);
    unsigned short* wkb = (unsigned short*)(ws + 10485760);
    unsigned short* wvb = (unsigned short*)(ws + 12582912);
    unsigned short* wob = (unsigned short*)(ws + 14680064);
    float* Qf = (float*)(ws + 16777216);
    float* Kf = (float*)(ws + 33554432);
    float* Vf = (float*)(ws + 50331648);
    unsigned short* Qbf = (unsigned short*)(ws + 67108864);
    unsigned short* Kbf = (unsigned short*)(ws + 75497472);
    unsigned short* Vt  = (unsigned short*)(ws + 83886080);
    unsigned short* attnb = (unsigned short*)(ws + 16777216);  // alias Qf

    const int M = B_DIM * S_DIM;   // 4096
    const int N = D_DIM;           // 1024
    const int K = D_DIM;           // 1024

    // 1) f32 -> bf16 operands
    cvt_f32_to_bf16<<<(M * K) / 256, 256, 0, stream>>>(x, xb, M * K);
    cvt_f32_to_bf16<<<(N * K) / 256, 256, 0, stream>>>(wq, wqb, N * K);
    cvt_f32_to_bf16<<<(N * K) / 256, 256, 0, stream>>>(wk, wkb, N * K);
    cvt_f32_to_bf16<<<(N * K) / 256, 256, 0, stream>>>(wv, wvb, N * K);
    cvt_f32_to_bf16<<<(N * K) / 256, 256, 0, stream>>>(wo, wob, N * K);

    // 2) QKV projections (WMMA)
    dim3 ggrid(N / 256, M / 64);
    gemm_bf16<<<ggrid, 256, 0, stream>>>(xb, wqb, bq, Qf, M, N, K);
    gemm_bf16<<<ggrid, 256, 0, stream>>>(xb, wkb, bk, Kf, M, N, K);
    gemm_bf16<<<ggrid, 256, 0, stream>>>(xb, wvb, bv, Vf, M, N, K);

    // 3) RoPE + head-major reshape (Q,K), V transpose
    rope_reshape<<<(B_DIM * S_DIM * H_DIM * 32) / 256, 256, 0, stream>>>(
        Qf, Kf, Qbf, Kbf);
    v_reshape<<<(B_DIM * S_DIM * D_DIM) / 256, 256, 0, stream>>>(Vf, Vt);

    // 4) Flash attention (async K/V staging + WMMA scores + WMMA PV)
    dim3 fgrid(S_DIM / 128, B_DIM * H_DIM);
    flash_attn<<<fgrid, 256, 0, stream>>>(Qbf, Kbf, Vt, am, attnb);

    // 5) Output projection (WMMA) -> f32 d_out
    gemm_bf16<<<ggrid, 256, 0, stream>>>(attnb, wob, bo, (float*)d_out, M, N, K);
}
